// SelfAttention_35966056136887
// MI455X (gfx1250) — matrix operand order repaired and checked by reading, not compiled, not run on hardware
//
#include <hip/hip_runtime.h>
#include <hip/hip_bf16.h>
#include <math.h>

// ---------------------------------------------------------------------------
// Self-attention for MI455X (gfx1250, wave32, WMMA bf16).
//   q = x Wq + bq ; k = x Wk + bk ; v = x Wv + bv
//   S = (q k^T / T) * mask ; P = softmax_rows(S) ; out = P v
// All matmuls run on v_wmma_f32_16x16x32_bf16 with fp32 accumulation.
// ---------------------------------------------------------------------------

typedef __attribute__((ext_vector_type(16))) __bf16 v16bf;
typedef __attribute__((ext_vector_type(8)))  __bf16 v8bf;
typedef __attribute__((ext_vector_type(8)))  float  v8f;

#define T_DIM 4096
#define D_DIM 1024

#define BM 128      // workgroup tile rows
#define BN 128      // workgroup tile cols
#define BK 32       // K step (one bf16 WMMA per staged chunk)
#define NTHREADS 256 // 8 wave32s: 2 (M) x 4 (N), each wave does 64x32

#define MODE_QKV    0   // C = A*B + bias, bf16 out
#define MODE_SCORES 1   // C = (A*B) * scale * mask, f32 out
#define MODE_PV     2   // C = A*B, f32 out

static __device__ __forceinline__ v8bf lds_ld8(const __bf16* p) {
    return *(const v8bf*)p;
}

static __device__ __forceinline__ void cvt16_f32_bf16(const float* f, v8bf& lo, v8bf& hi) {
#pragma unroll
    for (int i = 0; i < 8; ++i) {
        lo[i] = (__bf16)f[i];
        hi[i] = (__bf16)f[i + 8];
    }
}

template <int MODE, bool A_F32, bool B_F32, bool B_TRANS>
__global__ __launch_bounds__(NTHREADS)
void attn_gemm_wmma(const void* __restrict__ Ap,
                    const void* __restrict__ Bp,
                    void* __restrict__ Cp,
                    const float* __restrict__ bias,
                    const float* __restrict__ mask,
                    int M, int N, int K, float scale) {
    __shared__ __attribute__((aligned(16))) __bf16 lA[BM][BK + 8];  // +8 pad, rows stay 16B-aligned
    __shared__ __attribute__((aligned(16))) __bf16 lB[BK][BN + 8];

    const int tid  = threadIdx.x;
    const int lane = tid & 31;
    const int wave = tid >> 5;
    const int wm   = wave >> 2;   // 0..1  -> row offset wm*64
    const int wn   = wave & 3;    // 0..3  -> col offset wn*32

    const int m0 = blockIdx.y * BM;
    const int n0 = blockIdx.x * BN;

    // --- staging thread mappings ---------------------------------------
    const int ar = tid >> 1;            // A: 128 rows
    const int ah = (tid & 1) * 16;      // A: which 16-wide K half
    const int bkr = tid >> 3;           // B (no trans): 32 k-rows
    const int bseg = (tid & 7) * 16;    // B (no trans): 16-col segment
    const int btn = tid >> 1;           // B (trans): 128 source rows of K-matrix
    const int bth = (tid & 1) * 16;     // B (trans): 16-wide K half

    v8f acc[4][2];
#pragma unroll
    for (int mi = 0; mi < 4; ++mi)
#pragma unroll
        for (int ni = 0; ni < 2; ++ni)
            acc[mi][ni] = (v8f){0.f, 0.f, 0.f, 0.f, 0.f, 0.f, 0.f, 0.f};

    const int lrow = lane & 15;
    const int kb   = (lane < 16) ? 0 : 8;   // A-fragment K-octet select (ISA 7.12.2)

    for (int k0 = 0; k0 < K; k0 += BK) {
        // ---- stage A tile: [BM x BK] (fp32 -> bf16 convert on the fly) ----
        if (A_F32) {
            const float* src = (const float*)Ap + (size_t)(m0 + ar) * K + (k0 + ah);
            float buf[16];
#pragma unroll
            for (int i = 0; i < 4; ++i)
                ((float4*)buf)[i] = ((const float4*)src)[i];
            v8bf lo, hi;
            cvt16_f32_bf16(buf, lo, hi);
            *(v8bf*)&lA[ar][ah]     = lo;
            *(v8bf*)&lA[ar][ah + 8] = hi;
            if (k0 + BK < K) __builtin_prefetch(src + BK, 0, 1);
        } else {
            const __bf16* src = (const __bf16*)Ap + (size_t)(m0 + ar) * K + (k0 + ah);
            *(v8bf*)&lA[ar][ah]     = *(const v8bf*)src;
            *(v8bf*)&lA[ar][ah + 8] = *(const v8bf*)(src + 8);
            if (k0 + BK < K) __builtin_prefetch(src + BK, 0, 1);
        }

        // ---- stage B tile: [BK x BN] ----
        if (B_TRANS) {
            // Logical B[kk][n] = Kmat[n0+n][k0+kk] ; Kmat is row-major bf16 [N x K].
            const __bf16* src = (const __bf16*)Bp + (size_t)(n0 + btn) * K + (k0 + bth);
            v8bf lo = *(const v8bf*)src;
            v8bf hi = *(const v8bf*)(src + 8);
#pragma unroll
            for (int j = 0; j < 8; ++j) {
                lB[bth + j][btn]     = lo[j];
                lB[bth + 8 + j][btn] = hi[j];
            }
            if (k0 + BK < K) __builtin_prefetch(src + BK, 0, 1);
        } else if (B_F32) {
            const float* src = (const float*)Bp + (size_t)(k0 + bkr) * N + (n0 + bseg);
            float buf[16];
#pragma unroll
            for (int i = 0; i < 4; ++i)
                ((float4*)buf)[i] = ((const float4*)src)[i];
            v8bf lo, hi;
            cvt16_f32_bf16(buf, lo, hi);
            *(v8bf*)&lB[bkr][bseg]     = lo;
            *(v8bf*)&lB[bkr][bseg + 8] = hi;
            if (k0 + BK < K) __builtin_prefetch(src + (size_t)BK * N, 0, 1);
        } else {
            const __bf16* src = (const __bf16*)Bp + (size_t)(k0 + bkr) * N + (n0 + bseg);
            *(v8bf*)&lB[bkr][bseg]     = *(const v8bf*)src;
            *(v8bf*)&lB[bkr][bseg + 8] = *(const v8bf*)(src + 8);
            if (k0 + BK < K) __builtin_prefetch(src + (size_t)BK * N, 0, 1);
        }

        __syncthreads();

        // ---- load fragments per ISA 7.12.2 layouts ----
        v16bf afr[4];
#pragma unroll
        for (int mi = 0; mi < 4; ++mi) {
            const __bf16* row = &lA[wm * 64 + mi * 16 + lrow][0];
            v8bf lo = lds_ld8(row + kb);        // K = kb..kb+7
            v8bf hi = lds_ld8(row + kb + 16);   // K = kb+16..kb+23
            afr[mi] = __builtin_shufflevector(lo, hi,
                0, 1, 2, 3, 4, 5, 6, 7, 8, 9, 10, 11, 12, 13, 14, 15);
        }
        v16bf bfr[2];
#pragma unroll
        for (int ni = 0; ni < 2; ++ni) {
            const int col = wn * 32 + ni * 16 + lrow;  // lane%16 = N col, elements = K (table order)
#pragma unroll
            for (int j = 0; j < 8; ++j) {
                bfr[ni][j]     = lB[kb + j][col];
                bfr[ni][8 + j] = lB[kb + 16 + j][col];
            }
        }

        // ---- 8 WMMAs: D = A x B + C ----
#pragma unroll
        for (int mi = 0; mi < 4; ++mi)
#pragma unroll
            for (int ni = 0; ni < 2; ++ni)
                acc[mi][ni] = __builtin_amdgcn_wmma_f32_16x16x32_bf16(
                    false, afr[mi], false, bfr[ni], (short)0, acc[mi][ni],
                    false, false);

        __syncthreads();
    }

    // ---- epilogue: C/D layout lane = col (N), VGPR r = row r / r+8 ----
    const int col_in  = lane & 15;
    const int row_off = (lane < 16) ? 0 : 8;
#pragma unroll
    for (int mi = 0; mi < 4; ++mi) {
#pragma unroll
        for (int ni = 0; ni < 2; ++ni) {
            const int gm = m0 + wm * 64 + mi * 16 + row_off;
            const int gn = n0 + wn * 32 + ni * 16 + col_in;
#pragma unroll
            for (int r = 0; r < 8; ++r) {
                const size_t idx = (size_t)(gm + r) * N + gn;
                const float a = acc[mi][ni][r];
                if (MODE == MODE_QKV) {
                    ((__bf16*)Cp)[idx] = (__bf16)(a + bias[gn]);
                } else if (MODE == MODE_SCORES) {
                    ((float*)Cp)[idx] = a * scale * mask[idx];
                } else {
                    ((float*)Cp)[idx] = a;
                }
            }
        }
    }
}

// ---------------------------------------------------------------------------
// Row softmax over S [T_DIM x T_DIM], in place (fp32). One block per row.
// ---------------------------------------------------------------------------
__global__ __launch_bounds__(NTHREADS)
void attn_softmax_rows(float* __restrict__ S) {
    const int row = blockIdx.x;
    float* p = S + (size_t)row * T_DIM;
    const int tid = threadIdx.x;
    __shared__ float red[NTHREADS];

    // max
    float m = -INFINITY;
#pragma unroll
    for (int j = 0; j < T_DIM / NTHREADS; ++j)
        m = fmaxf(m, p[tid + j * NTHREADS]);
    red[tid] = m;
    __syncthreads();
    for (int s = NTHREADS / 2; s > 0; s >>= 1) {
        if (tid < s) red[tid] = fmaxf(red[tid], red[tid + s]);
        __syncthreads();
    }
    m = red[0];
    __syncthreads();

    // exp + sum
    float vals[T_DIM / NTHREADS];
    float sum = 0.f;
#pragma unroll
    for (int j = 0; j < T_DIM / NTHREADS; ++j) {
        vals[j] = __expf(p[tid + j * NTHREADS] - m);
        sum += vals[j];
    }
    red[tid] = sum;
    __syncthreads();
    for (int s = NTHREADS / 2; s > 0; s >>= 1) {
        if (tid < s) red[tid] += red[tid + s];
        __syncthreads();
    }
    const float inv = 1.f / red[0];

#pragma unroll
    for (int j = 0; j < T_DIM / NTHREADS; ++j)
        p[tid + j * NTHREADS] = vals[j] * inv;
}

// ---------------------------------------------------------------------------
extern "C" void kernel_launch(void* const* d_in, const int* in_sizes, int n_in,
                              void* d_out, int out_size, void* d_ws, size_t ws_size,
                              hipStream_t stream) {
    const float* input = (const float*)d_in[0];  // [T, D]
    const float* mask  = (const float*)d_in[1];  // [T, T]
    const float* Wq    = (const float*)d_in[2];  // [D, QK]
    const float* bq    = (const float*)d_in[3];
    const float* Wk    = (const float*)d_in[4];
    const float* bk    = (const float*)d_in[5];
    const float* Wv    = (const float*)d_in[6];
    const float* bv    = (const float*)d_in[7];
    float* out = (float*)d_out;                  // [T, V] fp32

    // workspace layout
    char* ws = (char*)d_ws;
    __bf16* Q  = (__bf16*)(ws);                          //  8 MB
    __bf16* Kb = (__bf16*)(ws + (size_t)8  * (1 << 20)); //  8 MB
    __bf16* Vb = (__bf16*)(ws + (size_t)16 * (1 << 20)); //  8 MB
    float*  S  = (float*) (ws + (size_t)24 * (1 << 20)); // 64 MB

    const dim3 blk(NTHREADS);

    // 1) Q/K/V projections: [4096x1024] = [4096x1024] x [1024x1024] + bias
    const dim3 gProj(D_DIM / BN, T_DIM / BM);
    attn_gemm_wmma<MODE_QKV, true, true, false><<<gProj, blk, 0, stream>>>(
        input, Wq, Q,  bq, nullptr, T_DIM, D_DIM, D_DIM, 1.0f);
    attn_gemm_wmma<MODE_QKV, true, true, false><<<gProj, blk, 0, stream>>>(
        input, Wk, Kb, bk, nullptr, T_DIM, D_DIM, D_DIM, 1.0f);
    attn_gemm_wmma<MODE_QKV, true, true, false><<<gProj, blk, 0, stream>>>(
        input, Wv, Vb, bv, nullptr, T_DIM, D_DIM, D_DIM, 1.0f);

    // 2) S = (Q K^T / T) * mask : [4096x4096]
    const dim3 gScores(T_DIM / BN, T_DIM / BM);
    attn_gemm_wmma<MODE_SCORES, false, false, true><<<gScores, blk, 0, stream>>>(
        Q, Kb, S, nullptr, mask, T_DIM, T_DIM, D_DIM, 1.0f / (float)T_DIM);

    // 3) row softmax in place
    attn_softmax_rows<<<T_DIM, blk, 0, stream>>>(S);

    // 4) out = P V : [4096x1024]
    attn_gemm_wmma<MODE_PV, true, false, false><<<gProj, blk, 0, stream>>>(
        S, Vb, out, nullptr, nullptr, T_DIM, D_DIM, T_DIM, 1.0f);
}
